// MultiHeadLinearAttention_67989332295684
// MI455X (gfx1250) — matrix operand order repaired and checked by reading, not compiled, not run on hardware
//
#include <hip/hip_runtime.h>
#include <hip/hip_bf16.h>
#include <math.h>

// ---------------------------------------------------------------------------
// MultiHeadLinearAttention for MI455X (gfx1250), bf16 WMMA path.
// M = B*S = 32768, K = N = 1024, H*D = 1024, OUT = 1024.
// ---------------------------------------------------------------------------

typedef __attribute__((ext_vector_type(16))) __bf16 v16bf;
typedef __attribute__((ext_vector_type(8)))  float  v8f;

union FragU {
    v16bf v;
    uint4 u[2];
};

#define KDIM   1024
#define NDIM   1024
#define MTOT   32768
#define SEQLEN 4096
#define NBATCH 8
#define MBLK_PER_B 32      // 4096 / 128

// -------------------- phase 1a: x fp32 -> bf16 -----------------------------
__global__ void cvt_x_bf16(const float* __restrict__ x,
                           __hip_bfloat16* __restrict__ out, int n) {
    int i = blockIdx.x * blockDim.x + threadIdx.x;
    if (i < n) out[i] = __float2bfloat16(x[i]);
}

// -------------------- phase 1b: weights -> bf16, transposed [N][K] ---------
__global__ void transpose_w_bf16(const float* __restrict__ w0,
                                 const float* __restrict__ w1,
                                 const float* __restrict__ w2,
                                 const float* __restrict__ w3,
                                 __hip_bfloat16* __restrict__ out) {
    __shared__ float tile[32][33];
    const float* src = (blockIdx.z == 0) ? w0 :
                       (blockIdx.z == 1) ? w1 :
                       (blockIdx.z == 2) ? w2 : w3;
    int k0 = blockIdx.x * 32;   // contraction index (input row)
    int n0 = blockIdx.y * 32;   // output-feature index (input col)
    int tx = threadIdx.x;       // 0..31
    int ty = threadIdx.y;       // 0..7
    for (int r = ty; r < 32; r += 8)
        tile[r][tx] = src[(size_t)(k0 + r) * NDIM + (n0 + tx)];
    __syncthreads();
    __hip_bfloat16* dst = out + (size_t)blockIdx.z * KDIM * NDIM;
    for (int r = ty; r < 32; r += 8)
        dst[(size_t)(n0 + r) * KDIM + (k0 + tx)] = __float2bfloat16(tile[tx][r]);
}

// -------------------- fragment loaders -------------------------------------
// A (16x32, row-major X, ld=KDIM): lane m = L&15, khalf = (L>>4)*8,
//   element j -> k = khalf + (j&7) + 16*(j>=8): two contiguous 16B chunks.
__device__ __forceinline__ v16bf load_fragA(const __hip_bfloat16* p) {
    FragU f;
    f.u[0] = *(const uint4*)(p);
    f.u[1] = *(const uint4*)(p + 16);
    return f.v;
}
// B (32x16, from Wt[n][k]): lane n = L&15, khalf = (L>>4)*8,
//   element j -> k = khalf + (j&7) + 16*(j>=8): two contiguous 16B chunks.
__device__ __forceinline__ v16bf load_fragB(const __hip_bfloat16* p) {
    FragU f;
    f.u[0] = *(const uint4*)(p);
    f.u[1] = *(const uint4*)(p + 16);
    return f.v;
}

__device__ __forceinline__ float feat_elu1(float x) {
    // elu(x)+1  ==  x>0 ? x+1 : exp(x)
    return x > 0.0f ? x + 1.0f : expf(x);
}

// -------------------- phase 2: fused QKV GEMM + reductions -----------------
// Grid: (MTOT/128, NDIM/64). Block: 256 threads = 8 waves laid out 4(M) x 2(N).
// Each wave: 2 M-tiles x 2 N-tiles x {q,k,v} = 12 f32 accumulators.
__global__ __launch_bounds__(256)
void qkv_wmma(const __hip_bfloat16* __restrict__ X,    // [32768,1024] bf16
              const __hip_bfloat16* __restrict__ Wt,   // [3][1024n][1024k] bf16
              const float* __restrict__ bq,
              const float* __restrict__ bk,
              const float* __restrict__ bv,
              __hip_bfloat16* __restrict__ Qout,       // [32768,1024] bf16
              float* __restrict__ KVpart,              // [8][32][1024]
              float* __restrict__ KSpart) {            // [8][32][1024]
    __shared__ float sKV[4][64];
    __shared__ float sKS[4][64];

    const int lane = threadIdx.x & 31;
    const int wave = threadIdx.x >> 5;
    const int mw = wave >> 1;           // 0..3
    const int nw = wave & 1;            // 0..1
    const int blockM = blockIdx.x * 128;
    const int blockN = blockIdx.y * 64;

    const int mA     = lane & 15;
    const int khalfA = (lane >> 4) * 8;
    const int nB     = lane & 15;
    const int khalfB = (lane >> 4) * 8;

    const __hip_bfloat16* Abase0 =
        X + (size_t)(blockM + mw * 32 + mA) * KDIM + khalfA;
    const __hip_bfloat16* Abase1 = Abase0 + (size_t)16 * KDIM;

    v8f acc[3][2][2] = {}; // [weight][mtile][ntile]

    for (int k = 0; k < KDIM; k += 32) {
        v16bf a0 = load_fragA(Abase0 + k);
        v16bf a1 = load_fragA(Abase1 + k);
#pragma unroll
        for (int w = 0; w < 3; ++w) {
#pragma unroll
            for (int nt = 0; nt < 2; ++nt) {
                const __hip_bfloat16* Bb =
                    Wt + (size_t)w * KDIM * NDIM +
                    (size_t)(blockN + nw * 32 + nt * 16 + nB) * KDIM +
                    khalfB + k;
                v16bf b = load_fragB(Bb);
                acc[w][0][nt] = __builtin_amdgcn_wmma_f32_16x16x32_bf16(
                    false, a0, false, b, (short)0, acc[w][0][nt], false, false);
                acc[w][1][nt] = __builtin_amdgcn_wmma_f32_16x16x32_bf16(
                    false, a1, false, b, (short)0, acc[w][1][nt], false, false);
            }
        }
    }

    // Epilogue: Q = elu(q)+1 stored; partials for KV = sum K*v, Ksum = sum K.
    float kvp[2] = {0.f, 0.f};
    float ksp[2] = {0.f, 0.f};
#pragma unroll
    for (int nt = 0; nt < 2; ++nt) {
        const int n = blockN + nw * 32 + nt * 16 + (lane & 15);
        const float bqv = bq[n], bkv = bk[n], bvv = bv[n];
#pragma unroll
        for (int mt = 0; mt < 2; ++mt) {
            const int mbase = blockM + mw * 32 + mt * 16 + (lane >> 4) * 8;
#pragma unroll
            for (int r = 0; r < 8; ++r) {
                float qv = acc[0][mt][nt][r] + bqv;
                Qout[(size_t)(mbase + r) * NDIM + n] =
                    __float2bfloat16(feat_elu1(qv));
                float Kf = feat_elu1(acc[1][mt][nt][r] + bkv);
                float Vf = acc[2][mt][nt][r] + bvv;
                kvp[nt] += Kf * Vf;
                ksp[nt] += Kf;
            }
        }
    }
    // Combine lane L with L^16 (same n, other 8-row half) -> 32-row sums.
#pragma unroll
    for (int nt = 0; nt < 2; ++nt) {
        kvp[nt] += __shfl_xor(kvp[nt], 16, 32);
        ksp[nt] += __shfl_xor(ksp[nt], 16, 32);
        if (lane < 16) {
            sKV[mw][nw * 32 + nt * 16 + lane] = kvp[nt];
            sKS[mw][nw * 32 + nt * 16 + lane] = ksp[nt];
        }
    }
    __syncthreads();
    if (threadIdx.x < 64) {
        const int nl = threadIdx.x;
        float s0 = sKV[0][nl] + sKV[1][nl] + sKV[2][nl] + sKV[3][nl];
        float s1 = sKS[0][nl] + sKS[1][nl] + sKS[2][nl] + sKS[3][nl];
        const int b = blockM >> 12;          // / SEQLEN
        const int j = blockIdx.x & (MBLK_PER_B - 1);
        const int n = blockN + nl;
        KVpart[((size_t)b * MBLK_PER_B + j) * NDIM + n] = s0;
        KSpart[((size_t)b * MBLK_PER_B + j) * NDIM + n] = s1;
    }
}

// -------------------- phase 3: deterministic partial reduction -------------
__global__ void reduce_kv(const float* __restrict__ KVpart,
                          const float* __restrict__ KSpart,
                          float* __restrict__ KV, float* __restrict__ KS) {
    int i = blockIdx.x * blockDim.x + threadIdx.x;   // 8192 = 8 * 1024
    if (i >= NBATCH * NDIM) return;
    int b = i >> 10, n = i & (NDIM - 1);
    float s0 = 0.f, s1 = 0.f;
    for (int j = 0; j < MBLK_PER_B; ++j) {
        s0 += KVpart[((size_t)b * MBLK_PER_B + j) * NDIM + n];
        s1 += KSpart[((size_t)b * MBLK_PER_B + j) * NDIM + n];
    }
    KV[i] = s0;
    KS[i] = s1;
}

// -------------------- phase 4: attn elementwise (in-place over Q) ----------
__global__ void attn_ew(__hip_bfloat16* __restrict__ Qbuf,
                        const float* __restrict__ KV,
                        const float* __restrict__ KS) {
    size_t i = (size_t)blockIdx.x * blockDim.x + threadIdx.x;
    int n = (int)(i & (NDIM - 1));
    int m = (int)(i >> 10);
    int b = m >> 12;                                // / SEQLEN
    float q = __bfloat162float(Qbuf[i]);
    float a = q * KV[b * NDIM + n] / (q * KS[b * NDIM + n] + 1e-6f);
    Qbuf[i] = __float2bfloat16(a);
}

// -------------------- phase 5: output GEMM + bias + GELU -------------------
__global__ __launch_bounds__(256)
void out_wmma(const __hip_bfloat16* __restrict__ A,   // attn bf16 [32768,1024]
              const __hip_bfloat16* __restrict__ Wt,  // wo_t bf16 [1024n][1024k]
              const float* __restrict__ bo,
              float* __restrict__ out) {
    const int lane = threadIdx.x & 31;
    const int wave = threadIdx.x >> 5;
    const int mw = wave >> 1;
    const int nw = wave & 1;
    const int blockM = blockIdx.x * 128;
    const int blockN = blockIdx.y * 64;

    const int mA     = lane & 15;
    const int khalfA = (lane >> 4) * 8;
    const int nB     = lane & 15;
    const int khalfB = (lane >> 4) * 8;

    const __hip_bfloat16* Abase0 =
        A + (size_t)(blockM + mw * 32 + mA) * KDIM + khalfA;
    const __hip_bfloat16* Abase1 = Abase0 + (size_t)16 * KDIM;

    v8f acc[2][2] = {};

    for (int k = 0; k < KDIM; k += 32) {
        v16bf a0 = load_fragA(Abase0 + k);
        v16bf a1 = load_fragA(Abase1 + k);
#pragma unroll
        for (int nt = 0; nt < 2; ++nt) {
            const __hip_bfloat16* Bb =
                Wt + (size_t)(blockN + nw * 32 + nt * 16 + nB) * KDIM +
                khalfB + k;
            v16bf b = load_fragB(Bb);
            acc[0][nt] = __builtin_amdgcn_wmma_f32_16x16x32_bf16(
                false, a0, false, b, (short)0, acc[0][nt], false, false);
            acc[1][nt] = __builtin_amdgcn_wmma_f32_16x16x32_bf16(
                false, a1, false, b, (short)0, acc[1][nt], false, false);
        }
    }

#pragma unroll
    for (int nt = 0; nt < 2; ++nt) {
        const int n = blockN + nw * 32 + nt * 16 + (lane & 15);
        const float bov = bo[n];
#pragma unroll
        for (int mt = 0; mt < 2; ++mt) {
            const int mbase = blockM + mw * 32 + mt * 16 + (lane >> 4) * 8;
#pragma unroll
            for (int r = 0; r < 8; ++r) {
                float y = acc[mt][nt][r] + bov;
                // tanh-approximate GELU (flax default)
                float t = tanhf(0.7978845608028654f *
                                (y + 0.044715f * y * y * y));
                out[(size_t)(mbase + r) * NDIM + n] = 0.5f * y * (1.0f + t);
            }
        }
    }
}

// ---------------------------------------------------------------------------
extern "C" void kernel_launch(void* const* d_in, const int* in_sizes, int n_in,
                              void* d_out, int out_size, void* d_ws,
                              size_t ws_size, hipStream_t stream) {
    const float* x  = (const float*)d_in[0];
    const float* wq = (const float*)d_in[1];
    const float* bq = (const float*)d_in[2];
    const float* wk = (const float*)d_in[3];
    const float* bk = (const float*)d_in[4];
    const float* wv = (const float*)d_in[5];
    const float* bv = (const float*)d_in[6];
    const float* wo = (const float*)d_in[7];
    const float* bo = (const float*)d_in[8];
    float* outp = (float*)d_out;

    char* ws = (char*)d_ws;
    // Workspace layout (bytes):
    const size_t OFF_X   = 0;                    // x bf16: 32768*1024*2 = 64MB
    const size_t OFF_W   = 67108864;             // 4 weights bf16^T: 8MB
    const size_t OFF_Q   = 75497472;             // Q/attn bf16 (in-place): 64MB
    const size_t OFF_KVP = 142606336;            // KV partials: 1MB
    const size_t OFF_KSP = 143654912;            // Ksum partials: 1MB
    const size_t OFF_KV  = 144703488;            // KV: 32KB
    const size_t OFF_KS  = 144736256;            // Ksum: 32KB

    __hip_bfloat16* Xb   = (__hip_bfloat16*)(ws + OFF_X);
    __hip_bfloat16* Wt   = (__hip_bfloat16*)(ws + OFF_W);
    __hip_bfloat16* Qb   = (__hip_bfloat16*)(ws + OFF_Q);
    float* KVpart        = (float*)(ws + OFF_KVP);
    float* KSpart        = (float*)(ws + OFF_KSP);
    float* KV            = (float*)(ws + OFF_KV);
    float* KS            = (float*)(ws + OFF_KS);

    const int nElem = MTOT * KDIM;               // 33,554,432

    // 1) convert activations & weights to bf16 (weights transposed to [N][K])
    cvt_x_bf16<<<nElem / 256, 256, 0, stream>>>(x, Xb, nElem);
    transpose_w_bf16<<<dim3(32, 32, 4), dim3(32, 8), 0, stream>>>(
        wq, wk, wv, wo, Wt);

    // 2) fused QKV GEMM + feature map + per-block sequence partials
    qkv_wmma<<<dim3(MTOT / 128, NDIM / 64), 256, 0, stream>>>(
        Xb, Wt, bq, bk, bv, Qb, KVpart, KSpart);

    // 3) deterministic reduction of partials -> KV[b,n], Ksum[b,n]
    reduce_kv<<<32, 256, 0, stream>>>(KVpart, KSpart, KV, KS);

    // 4) attn = Q*KV / (Q*Ksum + eps), in place (bf16)
    attn_ew<<<nElem / 256, 256, 0, stream>>>(Qb, KV, KS);

    // 5) output GEMM + bias + GELU -> fp32 out
    out_wmma<<<dim3(MTOT / 128, NDIM / 64), 256, 0, stream>>>(
        Qb, Wt + (size_t)3 * KDIM * NDIM, bo, outp);
}
